// MaskNet_29824252903645
// MI455X (gfx1250) — matrix operand order repaired and checked by reading, not compiled, not run on hardware
//
#include <hip/hip_runtime.h>

// ---- CDNA5 WMMA vector types -------------------------------------------------
typedef __attribute__((ext_vector_type(16))) _Float16 v16h;
typedef __attribute__((ext_vector_type(8)))  _Float16 h8;
typedef __attribute__((ext_vector_type(4)))  _Float16 h4;
typedef __attribute__((ext_vector_type(8)))  float    v8f;

#define IN_DIM          128     // per-node feature dim
#define KDIM            256     // 2*IN_DIM (concat features)
#define HIDDEN          64
#define TILE_E          16      // edges per WMMA M-tile
#define MTILES          2       // M-tiles processed together (share B frags)
#define WAVES_PER_BLOCK 8
#define ITERS_PER_WAVE  2       // outer iterations (32 edges each)

// ---- fp32 -> fp16 conversion (x_proj and w1, once per launch) ---------------
__global__ void cvt_f32_f16(const float* __restrict__ src,
                            _Float16* __restrict__ dst, int n4) {
  int i = blockIdx.x * blockDim.x + threadIdx.x;
  if (i < n4) {
    const float4 v = ((const float4*)src)[i];
    h4 h;
    h[0] = (_Float16)v.x; h[1] = (_Float16)v.y;
    h[2] = (_Float16)v.z; h[3] = (_Float16)v.w;
    ((h4*)dst)[i] = h;
  }
}

// ---- int64 edge ids -> u32 feature offsets (node * IN_DIM) ------------------
__global__ void prep_edges(const long long* __restrict__ eidx,
                           unsigned* __restrict__ ro,
                           unsigned* __restrict__ co, int E) {
  int i = blockIdx.x * blockDim.x + threadIdx.x;
  if (i < E) {
    ro[i] = (unsigned)eidx[i] * (unsigned)IN_DIM;
    co[i] = (unsigned)eidx[E + i] * (unsigned)IN_DIM;
  }
}

static __device__ __forceinline__ v16h load_frag(const _Float16* p) {
  h8 lo = *(const h8*)p;
  h8 hi = *(const h8*)(p + 8);
  return __builtin_shufflevector(lo, hi, 0, 1, 2, 3, 4, 5, 6, 7,
                                         8, 9, 10, 11, 12, 13, 14, 15);
}

// ---- main edge-MLP kernel ---------------------------------------------------
// Per 32-edge pair-tile: D[2x16x64] = A[2x16x256] x B[256x64]  (WMMA f16->f32)
// then out[e] = sigmoid(relu(D + b1) . w2 + b2)  (VALU epilogue)
__global__ void __launch_bounds__(256)
edge_mlp(const _Float16* __restrict__ xh,     // [N_NODES][128] f16
         const _Float16* __restrict__ w1h,    // [64][256]      f16 (row-major [out,in])
         const unsigned* __restrict__ ro,     // [E] row-node feature offsets
         const unsigned* __restrict__ co,     // [E] col-node feature offsets
         const float* __restrict__ b1,        // [64]
         const float* __restrict__ w2,        // [64]
         const float* __restrict__ b2p,       // [1]
         float* __restrict__ out,             // [E]
         int n_edges) {
  __shared__ _Float16 w1s[HIDDEN * KDIM];     // 32 KB of 320 KB WGP LDS

  // Cooperative stage of w1 (f16) into LDS: 2048 x 16B
  {
    const h8* __restrict__ s = (const h8*)w1h;
    h8* d = (h8*)w1s;
    for (int i = threadIdx.x; i < (HIDDEN * KDIM) / 8; i += blockDim.x)
      d[i] = s[i];
  }
  __syncthreads();

  const int lane = threadIdx.x & 31;
  const int m    = lane & 15;     // M row (A) / N col (B) handled by this lane
  const int hi   = lane >> 4;     // half-wave selector (K sub-range)
  const int waveInBlock = threadIdx.x >> 5;
  const int pair0 =
      (blockIdx.x * WAVES_PER_BLOCK + waveInBlock) * ITERS_PER_WAVE;

  // Per-lane constants for bias / layer-2 weights (N = nb*16 + m)
  const float bb[4]  = {b1[0 + m], b1[16 + m], b1[32 + m], b1[48 + m]};
  const float w2v[4] = {w2[0 + m], w2[16 + m], w2[32 + m], w2[48 + m]};
  const float bias2 = b2p[0];

  for (int t = 0; t < ITERS_PER_WAVE; ++t) {
    const int e_base = (pair0 + t) * (TILE_E * MTILES);
    if (e_base >= n_edges) break;            // wave-uniform guard (EXEC stays full)

    // Gather offsets: this lane's edge in each of the two M-tiles (u32 loads)
    const int e0 = e_base + m;
    const int e1 = e_base + TILE_E + m;
    const int c0 = (e0 < n_edges) ? e0 : (n_edges - 1);
    const int c1 = (e1 < n_edges) ? e1 : (n_edges - 1);

    // One 64-bit base pointer per node row (addr math done ONCE); all K-chunk
    // accesses below use compile-time offsets -> immediate-offset global loads.
    const _Float16* __restrict__ pr0 = xh + ro[c0] + hi * 8;
    const _Float16* __restrict__ pc0 = xh + co[c0] + hi * 8;
    const _Float16* __restrict__ pr1 = xh + ro[c1] + hi * 8;
    const _Float16* __restrict__ pc1 = xh + co[c1] + hi * 8;

    // Accumulators: 2 M-tiles x 4 N-blocks, zero-seeded (inline-0 C for WMMA)
    v8f acc0[4] = {}, acc1[4] = {};

#pragma unroll
    for (int kc = 0; kc < 8; ++kc) {         // K = 256 in chunks of 32
      const int k0 = kc * 32;
      // chunk lies fully in row-part (k<128) or col-part (k>=128)
      const _Float16* __restrict__ s0 = (k0 < IN_DIM) ? pr0 : pc0;
      const _Float16* __restrict__ s1 = (k0 < IN_DIM) ? pr1 : pc1;
      const int koff = k0 & (IN_DIM - 1);    // compile-time const per unrolled kc

      // A fragments (16-bit A 16x32 layout): lane m, K = {hi*8.., 16+hi*8..}
      h8 a0l = *(const h8*)(s0 + koff);
      h8 a0h = *(const h8*)(s0 + koff + 16);
      h8 a1l = *(const h8*)(s1 + koff);
      h8 a1h = *(const h8*)(s1 + koff + 16);
      v16h A0 = __builtin_shufflevector(a0l, a0h, 0, 1, 2, 3, 4, 5, 6, 7,
                                                  8, 9, 10, 11, 12, 13, 14, 15);
      v16h A1 = __builtin_shufflevector(a1l, a1h, 0, 1, 2, 3, 4, 5, 6, 7,
                                                  8, 9, 10, 11, 12, 13, 14, 15);

#pragma unroll
      for (int nb = 0; nb < 4; ++nb) {
        // B fragment: lane n = nb*16+m, K = {hi*8.., 16+hi*8..}
        const _Float16* bp = w1s + (nb * 16 + m) * KDIM + k0 + hi * 8;
        v16h B = __builtin_shufflevector(*(const h8*)bp, *(const h8*)(bp + 16),
                                         0, 1, 2, 3, 4, 5, 6, 7,
                                         8, 9, 10, 11, 12, 13, 14, 15);
        // Two independent WMMAs share this B fragment
        acc0[nb] = __builtin_amdgcn_wmma_f32_16x16x32_f16(
            false, A0, false, B, (short)0, acc0[nb], false, false);
        acc1[nb] = __builtin_amdgcn_wmma_f32_16x16x32_f16(
            false, A1, false, B, (short)0, acc1[nb], false, false);
      }
    }

    // Epilogue: +b1 -> relu -> dot(w2) -> reduce over N (16 lanes) -> sigmoid
#pragma unroll
    for (int mt = 0; mt < MTILES; ++mt) {
      const v8f* acc = (mt == 0) ? acc0 : acc1;
      const int tbase = e_base + mt * TILE_E;
#pragma unroll
      for (int r = 0; r < 8; ++r) {
        const float t0 = fmaxf(acc[0][r] + bb[0], 0.f);
        const float t1 = fmaxf(acc[1][r] + bb[1], 0.f);
        const float t2 = fmaxf(acc[2][r] + bb[2], 0.f);
        const float t3 = fmaxf(acc[3][r] + bb[3], 0.f);
        float v = t0 * w2v[0] + t1 * w2v[1] + t2 * w2v[2] + t3 * w2v[3];
        v += __shfl_xor(v, 1, 32);
        v += __shfl_xor(v, 2, 32);
        v += __shfl_xor(v, 4, 32);
        v += __shfl_xor(v, 8, 32);
        if ((lane & 15) == 0) {              // lane 0 -> rows 0..7, lane 16 -> 8..15
          const int eo = tbase + (hi << 3) + r;
          if (eo < n_edges)
            out[eo] = 1.0f / (1.0f + __expf(-(v + bias2)));
        }
      }
    }
  }
}

// ---- launch -----------------------------------------------------------------
extern "C" void kernel_launch(void* const* d_in, const int* in_sizes, int n_in,
                              void* d_out, int out_size, void* d_ws, size_t ws_size,
                              hipStream_t stream) {
  const float*     x    = (const float*)d_in[0];      // [N_NODES*128] f32
  const long long* eidx = (const long long*)d_in[1];  // [2*E] int64
  const float*     w1   = (const float*)d_in[2];      // [64*256] f32
  const float*     b1   = (const float*)d_in[3];      // [64]
  const float*     w2   = (const float*)d_in[4];      // [64]
  const float*     b2   = (const float*)d_in[5];      // [1]
  float*           out  = (float*)d_out;

  const int nx      = in_sizes[0];       // N_NODES * IN_DIM
  const int n_edges = in_sizes[1] / 2;
  const int nw      = in_sizes[2];       // HIDDEN * KDIM

  // Workspace layout (each region 256B-aligned):
  //   xh  : nx f16     | w1h : nw f16 | ro : E u32 | co : E u32
  char* wsp = (char*)d_ws;
  size_t off = 0;
  _Float16* xh  = (_Float16*)(wsp + off);
  off += ((size_t)nx * 2 + 255) & ~(size_t)255;
  _Float16* w1h = (_Float16*)(wsp + off);
  off += ((size_t)nw * 2 + 255) & ~(size_t)255;
  unsigned* ro  = (unsigned*)(wsp + off);
  off += ((size_t)n_edges * 4 + 255) & ~(size_t)255;
  unsigned* co  = (unsigned*)(wsp + off);

  {
    const int n4 = nx / 4;
    cvt_f32_f16<<<(n4 + 255) / 256, 256, 0, stream>>>(x, xh, n4);
    const int w4 = nw / 4;
    cvt_f32_f16<<<(w4 + 255) / 256, 256, 0, stream>>>(w1, w1h, w4);
    prep_edges<<<(n_edges + 255) / 256, 256, 0, stream>>>(eidx, ro, co, n_edges);
  }

  const long long edges_per_wave = (long long)TILE_E * MTILES * ITERS_PER_WAVE;
  const long long waves  = ((long long)n_edges + edges_per_wave - 1) / edges_per_wave;
  const int       blocks = (int)((waves + WAVES_PER_BLOCK - 1) / WAVES_PER_BLOCK);

  edge_mlp<<<blocks, WAVES_PER_BLOCK * 32, 0, stream>>>(
      xh, w1h, ro, co, b1, w2, b2, out, n_edges);
}
